// HierarchicalMinGRULayer_19318762897481
// MI455X (gfx1250) — compile-verified, matrix operand order checked by reading, not run on hardware
//
#include <hip/hip_runtime.h>
#include <hip/hip_bf16.h>
#include <math.h>

typedef __attribute__((ext_vector_type(16))) __bf16 v16bf;
typedef __attribute__((ext_vector_type(8)))  float  v8f;

#define B_SZ   4
#define S_SZ   4096
#define D_IN   1024
#define D_HALF 512
#define D_CAT  1024
#define NROWS  (B_SZ * S_SZ)     // 16384
#define TM     128               // 4 waves x 2 strips x 16 rows
#define TN     64
#define TK     32
#define NCHUNK 32
#define CHUNK  (S_SZ / NCHUNK)   // 128

// ---- A fragment: direct from global, bf16-converted in registers ------------
// CDNA5 16-bit A 16x32 layout: lanes 0-15 -> M=lane, K {0..7,16..23};
// lanes 16-31 -> M=lane-16, K {8..15,24..31}. Both K runs are 8 consecutive
// f32 in memory (K fast axis) -> 4x global_load_b128 per lane.
__device__ inline v16bf load_a_global(const float* __restrict__ X, int ldx,
                                      int row, int kbase) {
    const float* p = X + (size_t)row * ldx + kbase;
    const float4 f0 = *(const float4*)(p);
    const float4 f1 = *(const float4*)(p + 4);
    const float4 f2 = *(const float4*)(p + 16);
    const float4 f3 = *(const float4*)(p + 20);
    v16bf a;
    a[0]  = (__bf16)f0.x; a[1]  = (__bf16)f0.y; a[2]  = (__bf16)f0.z; a[3]  = (__bf16)f0.w;
    a[4]  = (__bf16)f1.x; a[5]  = (__bf16)f1.y; a[6]  = (__bf16)f1.z; a[7]  = (__bf16)f1.w;
    a[8]  = (__bf16)f2.x; a[9]  = (__bf16)f2.y; a[10] = (__bf16)f2.z; a[11] = (__bf16)f2.w;
    a[12] = (__bf16)f3.x; a[13] = (__bf16)f3.y; a[14] = (__bf16)f3.z; a[15] = (__bf16)f3.w;
    return a;
}

// ---- B fragment gather: one fragment slot per thread, packed v16bf ----------
// B 32x16 layout: lanes 0-15 -> N=lane, K {0..7,16..23}; lanes 16-31 -> {8..15,24..31}
__device__ inline v16bf gather_b_global(const float* __restrict__ W, int ldw,
                                        int k0, int kh, int col) {
    v16bf b;
#pragma unroll
    for (int i = 0; i < 8; ++i) {
        b[i]     = (__bf16)W[(size_t)(k0 + kh + i) * ldw + col];
        b[i + 8] = (__bf16)W[(size_t)(k0 + kh + 16 + i) * ldw + col];
    }
    return b;
}

// ---- Projection GEMM: z/h projections + sigmoid fusion -> scan coeffs -------
//   zp = x@Wz + bz ; hp = x@Wh + bh ; s = sigmoid(zp)
//   a_out = 1 - s ; b_out = s * hp      (coeffs of h_t = a_t h_{t-1} + b_t)
__global__ __launch_bounds__(128) void proj_kernel(
    const float* __restrict__ x,
    const float* __restrict__ Wz, const float* __restrict__ bz,
    const float* __restrict__ Wh, const float* __restrict__ bh,
    float* __restrict__ a_out, float* __restrict__ b_out, int comp)
{
    // fragment-ready LDS: slot t = (col-group nt = t>>5, lane = t&31), 16 halves each
    __shared__ __attribute__((aligned(32))) __bf16 Bz_s[TK * TN];
    __shared__ __attribute__((aligned(32))) __bf16 Bh_s[TK * TN];

    const int rowBase = blockIdx.x * TM;
    const int nBase   = blockIdx.y * TN;
    const int tid  = threadIdx.x;
    const int wave = tid >> 5;
    const int lane = tid & 31;

    // this thread's B-fill slot
    const int f_nt  = tid >> 5;
    const int f_col = nBase + f_nt * 16 + (lane & 15);
    const int f_kh  = (lane >> 4) << 3;       // 0 or 8
    // this lane's A-fragment rows (two 16-row strips per wave)
    const int aRow0 = rowBase + wave * 32 + (lane & 15);
    const int aRow1 = aRow0 + 16;
    const int aKh   = (lane >> 4) << 3;

    v8f accz[2][4], acch[2][4];
    const v8f vzero = {0.f, 0.f, 0.f, 0.f, 0.f, 0.f, 0.f, 0.f};
#pragma unroll
    for (int s = 0; s < 2; ++s)
#pragma unroll
        for (int i = 0; i < 4; ++i) { accz[s][i] = vzero; acch[s][i] = vzero; }

    for (int k0 = 0; k0 < D_IN; k0 += TK) {
        const v16bf vz = gather_b_global(Wz, D_HALF, k0, f_kh, f_col);
        const v16bf vh = gather_b_global(Wh, D_HALF, k0, f_kh, f_col);
        __syncthreads();                      // previous iter's frag reads done
        *(v16bf*)&Bz_s[tid * 16] = vz;
        *(v16bf*)&Bh_s[tid * 16] = vh;
        const v16bf af0 = load_a_global(x, D_IN, aRow0, k0 + aKh);
        const v16bf af1 = load_a_global(x, D_IN, aRow1, k0 + aKh);
        __syncthreads();                      // tiles visible
#pragma unroll
        for (int nt = 0; nt < 4; ++nt) {
            const v16bf bzf = *(const v16bf*)&Bz_s[(nt * 32 + lane) * 16];
            const v16bf bhf = *(const v16bf*)&Bh_s[(nt * 32 + lane) * 16];
            accz[0][nt] = __builtin_amdgcn_wmma_f32_16x16x32_bf16(
                false, af0, false, bzf, (short)0, accz[0][nt], false, false);
            accz[1][nt] = __builtin_amdgcn_wmma_f32_16x16x32_bf16(
                false, af1, false, bzf, (short)0, accz[1][nt], false, false);
            acch[0][nt] = __builtin_amdgcn_wmma_f32_16x16x32_bf16(
                false, af0, false, bhf, (short)0, acch[0][nt], false, false);
            acch[1][nt] = __builtin_amdgcn_wmma_f32_16x16x32_bf16(
                false, af1, false, bhf, (short)0, acch[1][nt], false, false);
        }
    }

    // epilogue: C/D layout -> lanes 0-15 rows r, lanes 16-31 rows r+8; col = lane&15
    const int rsel = (lane >> 4) << 3;
    const int coll = lane & 15;
#pragma unroll
    for (int s = 0; s < 2; ++s) {
        const int m0 = rowBase + wave * 32 + s * 16 + rsel;
#pragma unroll
        for (int nt = 0; nt < 4; ++nt) {
            const int col  = nBase + nt * 16 + coll;
            const float bZ = bz[col];
            const float bH = bh[col];
#pragma unroll
            for (int r = 0; r < 8; ++r) {
                const float zp = accz[s][nt][r] + bZ;
                const float hp = acch[s][nt][r] + bH;
                const float sg = 1.0f / (1.0f + expf(-zp));
                const size_t idx = (size_t)(m0 + r) * D_CAT + comp * D_HALF + col;
                a_out[idx] = 1.0f - sg;
                b_out[idx] = sg * hp;
            }
        }
    }
}

// ---- Chunked linear scan: h_t = a_t h_{t-1} + b_t over S, per (batch,channel)
__global__ __launch_bounds__(256) void scan_local_kernel(
    const float* __restrict__ a_buf, const float* __restrict__ b_buf,
    float* __restrict__ Ach, float* __restrict__ Hch)
{
    const int g  = blockIdx.x * 256 + threadIdx.x;   // B*NCHUNK*D_CAT threads
    const int c  = g & (D_CAT - 1);
    const int ch = (g >> 10) & (NCHUNK - 1);
    const int bi = g >> 15;
    float A = 1.0f, H = 0.0f;
    const int s0 = ch * CHUNK;
    for (int t = 0; t < CHUNK; ++t) {
        const size_t idx = ((size_t)(bi * S_SZ + s0 + t)) * D_CAT + c;
        const float av = a_buf[idx];
        A *= av;
        H = av * H + b_buf[idx];
    }
    const size_t o = (size_t)ch * (B_SZ * D_CAT) + bi * D_CAT + c;
    Ach[o] = A;
    Hch[o] = H;
}

__global__ __launch_bounds__(256) void scan_carry_kernel(
    const float* __restrict__ Ach, const float* __restrict__ Hch,
    float* __restrict__ cin)
{
    const int g  = blockIdx.x * 256 + threadIdx.x;   // B*D_CAT threads
    const int c  = g & (D_CAT - 1);
    const int bi = g >> 10;
    float carry = 0.0f;
    for (int j = 0; j < NCHUNK; ++j) {
        const size_t o = (size_t)j * (B_SZ * D_CAT) + bi * D_CAT + c;
        cin[o] = carry;
        carry = Ach[o] * carry + Hch[o];
    }
}

__global__ __launch_bounds__(256) void scan_apply_kernel(
    const float* __restrict__ a_buf, const float* __restrict__ b_buf,
    const float* __restrict__ cin, float* __restrict__ hc)
{
    const int g  = blockIdx.x * 256 + threadIdx.x;
    const int c  = g & (D_CAT - 1);
    const int ch = (g >> 10) & (NCHUNK - 1);
    const int bi = g >> 15;
    float h = cin[(size_t)ch * (B_SZ * D_CAT) + bi * D_CAT + c];
    const int s0 = ch * CHUNK;
    for (int t = 0; t < CHUNK; ++t) {
        const size_t idx = ((size_t)(bi * S_SZ + s0 + t)) * D_CAT + c;
        h = a_buf[idx] * h + b_buf[idx];
        hc[idx] = h;
    }
}

// ---- Gate GEMM: out = sigmoid(hc @ Wg + bg) * hc ----------------------------
__global__ __launch_bounds__(128) void gate_kernel(
    const float* __restrict__ hc,
    const float* __restrict__ Wg, const float* __restrict__ bg,
    float* __restrict__ out)
{
    __shared__ __attribute__((aligned(32))) __bf16 Bs_s[TK * TN];

    const int rowBase = blockIdx.x * TM;
    const int nBase   = blockIdx.y * TN;
    const int tid  = threadIdx.x;
    const int wave = tid >> 5;
    const int lane = tid & 31;

    const int f_nt  = tid >> 5;
    const int f_col = nBase + f_nt * 16 + (lane & 15);
    const int f_kh  = (lane >> 4) << 3;
    const int aRow0 = rowBase + wave * 32 + (lane & 15);
    const int aRow1 = aRow0 + 16;
    const int aKh   = (lane >> 4) << 3;

    v8f acc[2][4];
    const v8f vzero = {0.f, 0.f, 0.f, 0.f, 0.f, 0.f, 0.f, 0.f};
#pragma unroll
    for (int s = 0; s < 2; ++s)
#pragma unroll
        for (int i = 0; i < 4; ++i) acc[s][i] = vzero;

    for (int k0 = 0; k0 < D_CAT; k0 += TK) {
        const v16bf vg = gather_b_global(Wg, D_CAT, k0, f_kh, f_col);
        __syncthreads();
        *(v16bf*)&Bs_s[tid * 16] = vg;
        const v16bf af0 = load_a_global(hc, D_CAT, aRow0, k0 + aKh);
        const v16bf af1 = load_a_global(hc, D_CAT, aRow1, k0 + aKh);
        __syncthreads();
#pragma unroll
        for (int nt = 0; nt < 4; ++nt) {
            const v16bf bf = *(const v16bf*)&Bs_s[(nt * 32 + lane) * 16];
            acc[0][nt] = __builtin_amdgcn_wmma_f32_16x16x32_bf16(
                false, af0, false, bf, (short)0, acc[0][nt], false, false);
            acc[1][nt] = __builtin_amdgcn_wmma_f32_16x16x32_bf16(
                false, af1, false, bf, (short)0, acc[1][nt], false, false);
        }
    }

    const int rsel = (lane >> 4) << 3;
    const int coll = lane & 15;
#pragma unroll
    for (int s = 0; s < 2; ++s) {
        const int m0 = rowBase + wave * 32 + s * 16 + rsel;
#pragma unroll
        for (int nt = 0; nt < 4; ++nt) {
            const int col  = nBase + nt * 16 + coll;
            const float bG = bg[col];
#pragma unroll
            for (int r = 0; r < 8; ++r) {
                const float gp = acc[s][nt][r] + bG;
                const float sg = 1.0f / (1.0f + expf(-gp));
                const size_t idx = (size_t)(m0 + r) * D_CAT + col;
                out[idx] = sg * hc[idx];
            }
        }
    }
}

// ---- Host-side launch --------------------------------------------------------
extern "C" void kernel_launch(void* const* d_in, const int* in_sizes, int n_in,
                              void* d_out, int out_size, void* d_ws, size_t ws_size,
                              hipStream_t stream) {
    (void)in_sizes; (void)n_in; (void)out_size; (void)ws_size;
    const float* x    = (const float*)d_in[0];
    const float* W_zf = (const float*)d_in[1];
    const float* b_zf = (const float*)d_in[2];
    const float* W_hf = (const float*)d_in[3];
    const float* b_hf = (const float*)d_in[4];
    const float* W_zs = (const float*)d_in[5];
    const float* b_zs = (const float*)d_in[6];
    const float* W_hs = (const float*)d_in[7];
    const float* b_hs = (const float*)d_in[8];
    const float* W_g  = (const float*)d_in[9];
    const float* b_g  = (const float*)d_in[10];
    float* out = (float*)d_out;

    float* a_buf = (float*)d_ws;                          // NROWS*D_CAT
    float* b_buf = a_buf + (size_t)NROWS * D_CAT;
    float* hc    = b_buf + (size_t)NROWS * D_CAT;
    float* Ach   = hc    + (size_t)NROWS * D_CAT;
    float* Hch   = Ach   + (size_t)NCHUNK * B_SZ * D_CAT;
    float* cin   = Hch   + (size_t)NCHUNK * B_SZ * D_CAT;

    const dim3 projGrid(NROWS / TM, D_HALF / TN);
    proj_kernel<<<projGrid, 128, 0, stream>>>(x, W_zf, b_zf, W_hf, b_hf,
                                              a_buf, b_buf, 0);
    proj_kernel<<<projGrid, 128, 0, stream>>>(x, W_zs, b_zs, W_hs, b_hs,
                                              a_buf, b_buf, 1);

    const int scanThreads = B_SZ * NCHUNK * D_CAT;        // 131072
    scan_local_kernel<<<scanThreads / 256, 256, 0, stream>>>(a_buf, b_buf, Ach, Hch);
    scan_carry_kernel<<<(B_SZ * D_CAT) / 256, 256, 0, stream>>>(Ach, Hch, cin);
    scan_apply_kernel<<<scanThreads / 256, 256, 0, stream>>>(a_buf, b_buf, cin, hc);

    const dim3 gateGrid(NROWS / TM, D_CAT / TN);
    gate_kernel<<<gateGrid, 128, 0, stream>>>(hc, W_g, b_g, out);
}
